// EB_936302870589
// MI455X (gfx1250) — matrix operand order repaired and checked by reading, not compiled, not run on hardware
//
#include <hip/hip_runtime.h>
#include <math.h>

typedef __attribute__((ext_vector_type(16))) _Float16 v16h;
typedef __attribute__((ext_vector_type(8)))  float    v8f;

#define NN 100000
#define NE 1600000

// ---- parameter blob offsets (floats), dict insertion order: W0,b0,[g0,bt0],W1,b1,[g1,bt1],... ----
// phi_e : dims [66,32,32]
#define PE_W0  0
#define PE_B0  2112
#define PE_G0  2144
#define PE_BT0 2176
#define PE_W1  2208
#define PE_B1  3232
// phi_x : dims [32,32,32,1]
#define PX_W0  0
#define PX_B0  1024
#define PX_G0  1056
#define PX_BT0 1088
#define PX_W1  1120
#define PX_B1  2144
#define PX_G1  2176
#define PX_BT1 2208
#define PX_W2  2240
#define PX_B2  2272
// phi_m : dims [32,32,32] ln_out
#define PM_W0  0
#define PM_B0  1024
#define PM_G0  1056
#define PM_BT0 1088
#define PM_W1  1120
#define PM_B1  2144
#define PM_G1  2176
#define PM_BT1 2208
// phi_s : dims [64,32,32]
#define PS_W0  0
#define PS_B0  2048
#define PS_G0  2080
#define PS_BT0 2112
#define PS_W1  2144
#define PS_B1  3168

// ---- workspace layout (bytes) ----
#define WPK_BYTES   (20 * 1024)        // 20 packed f16 A-tiles, 1KB each
#define XAGG_OFF    65536
#define SAGG_OFF    (65536 + NN * 3 * 4)

__device__ inline v8f vzero() { v8f z = {}; return z; }

__device__ inline v8f wmma_f16(v16h a, v16h b, v8f c) {
  return __builtin_amdgcn_wmma_f32_16x16x32_f16(false, a, false, b, (short)0, c,
                                                false, false);
}

__device__ inline float psi_f(float x) { return copysignf(log1pf(fabsf(x)), x); }

// Load one pre-packed 16x32 f16 A-tile: 8 coalesced dwords per lane.
__device__ inline v16h load_tileA(const unsigned int* __restrict__ wpk, int tile, int lane) {
  union { v16h h; unsigned int u[8]; } t;
  const unsigned int* p = wpk + tile * 256 + lane;
#pragma unroll
  for (int vv = 0; vv < 8; ++vv) t.u[vv] = p[vv * 32];
  return t.h;
}

// Build B operand (32K x 16N, f16) from 16 f32 features of this lane's K-half.
__device__ inline v16h pack_b16(const float* f) {
  v16h h;
#pragma unroll
  for (int k = 0; k < 16; ++k) h[k] = (_Float16)f[k];
  return h;
}

// Convert f32 D-layout (two 16x16 tiles: features 0-15 / 16-31) to f16 B-layout.
__device__ inline v16h d_to_b(v8f t0, v8f t1, int lane) {
  float f[16];
  bool lo = (lane & 16) == 0;
#pragma unroll
  for (int r = 0; r < 8; ++r) {
    float o0 = __shfl_xor(t0[r], 16, 32);
    float o1 = __shfl_xor(t1[r], 16, 32);
    f[r]     = lo ? t0[r] : o1;   // grp0: feat r      ; grp1: feat 16+r
    f[8 + r] = lo ? o0    : t1[r];// grp0: feat 8+r    ; grp1: feat 24+r
  }
  return pack_b16(f);
}

__device__ inline void add_bias(v8f& t0, v8f& t1, const float* __restrict__ b, int base) {
#pragma unroll
  for (int r = 0; r < 8; ++r) { t0[r] += b[base + r]; t1[r] += b[16 + base + r]; }
}

// LayerNorm over 32 features (D-layout) + ReLU or sigmoid, f32 math.
__device__ inline void ln_act(v8f& t0, v8f& t1, const float* __restrict__ g,
                              const float* __restrict__ bt, int base, bool relu) {
  float sm = 0.f, sq = 0.f;
#pragma unroll
  for (int r = 0; r < 8; ++r) {
    sm += t0[r] + t1[r];
    sq += t0[r] * t0[r] + t1[r] * t1[r];
  }
  sm += __shfl_xor(sm, 16, 32);
  sq += __shfl_xor(sq, 16, 32);
  float mean = sm * (1.0f / 32.0f);
  float var  = sq * (1.0f / 32.0f) - mean * mean;
  float rstd = rsqrtf(var + 1e-5f);
#pragma unroll
  for (int r = 0; r < 8; ++r) {
    float x0 = (t0[r] - mean) * rstd * g[base + r] + bt[base + r];
    float x1 = (t1[r] - mean) * rstd * g[16 + base + r] + bt[16 + base + r];
    if (relu) { x0 = fmaxf(x0, 0.f); x1 = fmaxf(x1, 0.f); }
    else      { x0 = 1.f / (1.f + __expf(-x0)); x1 = 1.f / (1.f + __expf(-x1)); }
    t0[r] = x0; t1[r] = x1;
  }
}

// ---- kernel 0: zero the accumulators ----
__global__ void EB_zero_kernel(float* __restrict__ p, long long count) {
  long long idx    = (long long)blockIdx.x * blockDim.x + threadIdx.x;
  long long stride = (long long)gridDim.x * blockDim.x;
  for (; idx < count; idx += stride) p[idx] = 0.f;
}

// ---- kernel 1: pack all weight matrices into f16 A-matrix layout ----
// Packed A order (contiguous k, as in B): lane = grp*16+m (m=out-row); VGPR v holds K pair
// kb(v,grp) = 16*grp + 2v.
__global__ void EB_pack_kernel(const float* __restrict__ pe, const float* __restrict__ px,
                               const float* __restrict__ pm, const float* __restrict__ ps,
                               unsigned int* __restrict__ wpk) {
  int t = threadIdx.x;
  for (int it = 0; it < 20; ++it) {
    int tile = it;
    int rem  = t;               // 256 entries per tile, one per thread
    int vv   = rem >> 5;
    int lane = rem & 31;
    int grp  = lane >> 4;
    int m    = lane & 15;
    int kb   = 16 * grp + 2 * vv;
    const float* W = pe; int k0 = 0, m0 = 0;
    switch (tile) {
      case 0:  W = pe + PE_W0; k0 = 2;  m0 = 0;  break;  // phi_e W0 rows for s_i
      case 1:  W = pe + PE_W0; k0 = 2;  m0 = 16; break;
      case 2:  W = pe + PE_W0; k0 = 34; m0 = 0;  break;  // phi_e W0 rows for s_j
      case 3:  W = pe + PE_W0; k0 = 34; m0 = 16; break;
      case 4:  W = pe + PE_W1; m0 = 0;  break;
      case 5:  W = pe + PE_W1; m0 = 16; break;
      case 6:  W = pm + PM_W0; m0 = 0;  break;
      case 7:  W = pm + PM_W0; m0 = 16; break;
      case 8:  W = pm + PM_W1; m0 = 0;  break;
      case 9:  W = pm + PM_W1; m0 = 16; break;
      case 10: W = px + PX_W0; m0 = 0;  break;
      case 11: W = px + PX_W0; m0 = 16; break;
      case 12: W = px + PX_W1; m0 = 0;  break;
      case 13: W = px + PX_W1; m0 = 16; break;
      case 14: W = ps + PS_W0; k0 = 0;  m0 = 0;  break;  // phi_s W0 rows for s
      case 15: W = ps + PS_W0; k0 = 0;  m0 = 16; break;
      case 16: W = ps + PS_W0; k0 = 32; m0 = 0;  break;  // phi_s W0 rows for s_agg
      case 17: W = ps + PS_W0; k0 = 32; m0 = 16; break;
      case 18: W = ps + PS_W1; m0 = 0;  break;
      default: W = ps + PS_W1; m0 = 16; break;
    }
    float lo = W[(k0 + kb)     * 32 + m0 + m];
    float hi = W[(k0 + kb + 1) * 32 + m0 + m];
    union { _Float16 h[2]; unsigned int u; } pk;
    pk.h[0] = (_Float16)lo; pk.h[1] = (_Float16)hi;
    wpk[tile * 256 + vv * 32 + lane] = pk.u;
  }
}

// ---- kernel 2: per-edge pipeline. One wave = 16 edges. ----
__global__ __launch_bounds__(256) void EB_edge_kernel(
    const float* __restrict__ v, const float* __restrict__ s,
    const float* __restrict__ pe, const float* __restrict__ px,
    const float* __restrict__ pm, const long long* __restrict__ ei,
    const unsigned int* __restrict__ wpk, float* __restrict__ e_out,
    float* __restrict__ x_agg, float* __restrict__ s_agg) {
  int lane = threadIdx.x & 31;
  int wave = threadIdx.x >> 5;
  int n    = lane & 15;
  int grp  = lane >> 4;
  int base = grp * 8;
  long long e = (long long)(blockIdx.x * 8 + wave) * 16 + n;
  long long i = ei[e];
  long long j = ei[(long long)NE + e];

  // geometry features (duplicated on both lane groups)
  float vi0 = v[i * 3 + 0], vi1 = v[i * 3 + 1], vi2 = v[i * 3 + 2];
  float vj0 = v[j * 3 + 0], vj1 = v[j * 3 + 1], vj2 = v[j * 3 + 2];
  float d0 = vi0 - vj0, d1 = vi1 - vj1, d2 = vi2 - vj2;
  float nrm = psi_f(d0 * d0 + d1 * d1 + d2 * d2);
  float dot = psi_f(vi0 * vj0 + vi1 * vj1 + vi2 * vj2);

  // B operands from gathered s rows (lane group = K-half)
  float fi[16], fj[16];
  const float* si = s + i * 32 + grp * 16;
  const float* sj = s + j * 32 + grp * 16;
#pragma unroll
  for (int k = 0; k < 16; ++k) { fi[k] = si[k]; fj[k] = sj[k]; }
  v16h Bi = pack_b16(fi), Bj = pack_b16(fj);

  // phi_e layer0: 66->32  (WMMA over s_i, s_j blocks + rank-2 geometry update)
  v8f t0 = vzero(), t1 = vzero();
  t0 = wmma_f16(load_tileA(wpk, 0, lane), Bi, t0);
  t1 = wmma_f16(load_tileA(wpk, 1, lane), Bi, t1);
  t0 = wmma_f16(load_tileA(wpk, 2, lane), Bj, t0);
  t1 = wmma_f16(load_tileA(wpk, 3, lane), Bj, t1);
#pragma unroll
  for (int r = 0; r < 8; ++r) {
    int m0i = base + r, m1i = 16 + base + r;
    t0[r] += pe[PE_W0 + m0i] * nrm + pe[PE_W0 + 32 + m0i] * dot + pe[PE_B0 + m0i];
    t1[r] += pe[PE_W0 + m1i] * nrm + pe[PE_W0 + 32 + m1i] * dot + pe[PE_B0 + m1i];
  }
  ln_act(t0, t1, pe + PE_G0, pe + PE_BT0, base, true);

  // phi_e layer1: 32->32 (linear) -> e_ij
  v16h Bx = d_to_b(t0, t1, lane);
  v8f u0 = vzero(), u1 = vzero();
  u0 = wmma_f16(load_tileA(wpk, 4, lane), Bx, u0);
  u1 = wmma_f16(load_tileA(wpk, 5, lane), Bx, u1);
  add_bias(u0, u1, pe + PE_B1, base);
  float* ep = e_out + e * 32;
#pragma unroll
  for (int r = 0; r < 8; ++r) { ep[base + r] = u0[r]; ep[16 + base + r] = u1[r]; }

  // phi_m: 32->32 (LN+relu) -> 32->32 (LN+sigmoid) = m_ij
  Bx = d_to_b(u0, u1, lane);
  v8f a0 = vzero(), a1 = vzero();
  a0 = wmma_f16(load_tileA(wpk, 6, lane), Bx, a0);
  a1 = wmma_f16(load_tileA(wpk, 7, lane), Bx, a1);
  add_bias(a0, a1, pm + PM_B0, base);
  ln_act(a0, a1, pm + PM_G0, pm + PM_BT0, base, true);
  Bx = d_to_b(a0, a1, lane);
  v8f m0v = vzero(), m1v = vzero();
  m0v = wmma_f16(load_tileA(wpk, 8, lane), Bx, m0v);
  m1v = wmma_f16(load_tileA(wpk, 9, lane), Bx, m1v);
  add_bias(m0v, m1v, pm + PM_B1, base);
  ln_act(m0v, m1v, pm + PM_G1, pm + PM_BT1, base, false);  // sigmoid

  // scatter m_ij into s_agg[i]
  float* sa = s_agg + i * 32;
#pragma unroll
  for (int r = 0; r < 8; ++r) {
    atomicAdd(sa + base + r, m0v[r]);
    atomicAdd(sa + 16 + base + r, m1v[r]);
  }

  // phi_x: 32->32 (LN+relu) -> 32->32 (LN+relu) -> 32->1
  Bx = d_to_b(m0v, m1v, lane);
  v8f h0 = vzero(), h1 = vzero();
  h0 = wmma_f16(load_tileA(wpk, 10, lane), Bx, h0);
  h1 = wmma_f16(load_tileA(wpk, 11, lane), Bx, h1);
  add_bias(h0, h1, px + PX_B0, base);
  ln_act(h0, h1, px + PX_G0, px + PX_BT0, base, true);
  Bx = d_to_b(h0, h1, lane);
  v8f q0 = vzero(), q1 = vzero();
  q0 = wmma_f16(load_tileA(wpk, 12, lane), Bx, q0);
  q1 = wmma_f16(load_tileA(wpk, 13, lane), Bx, q1);
  add_bias(q0, q1, px + PX_B1, base);
  ln_act(q0, q1, px + PX_G1, px + PX_BT1, base, true);
  float p = 0.f;
#pragma unroll
  for (int r = 0; r < 8; ++r)
    p += q0[r] * px[PX_W2 + base + r] + q1[r] * px[PX_W2 + 16 + base + r];
  p += __shfl_xor(p, 16, 32);
  float scal = p + px[PX_B2];

  // coordinate update scatter (one lane per edge)
  if (lane < 16) {
    float w0 = fminf(fmaxf(d0 * scal, -100.f), 100.f);
    float w1 = fminf(fmaxf(d1 * scal, -100.f), 100.f);
    float w2 = fminf(fmaxf(d2 * scal, -100.f), 100.f);
    atomicAdd(x_agg + i * 3 + 0, w0);
    atomicAdd(x_agg + i * 3 + 1, w1);
    atomicAdd(x_agg + i * 3 + 2, w2);
  }
}

// ---- kernel 3: per-node update. One wave = 16 nodes. ----
__global__ __launch_bounds__(256) void EB_node_kernel(
    const float* __restrict__ v, const float* __restrict__ s,
    const float* __restrict__ ps, const unsigned int* __restrict__ wpk,
    const float* __restrict__ x_agg, const float* __restrict__ s_agg,
    float* __restrict__ out_v, float* __restrict__ out_s) {
  int lane = threadIdx.x & 31;
  int wave = threadIdx.x >> 5;
  long long nb = (long long)(blockIdx.x * 8 + wave) * 16;
  if (nb >= NN) return;  // uniform per wave: EXEC stays all-ones for live waves
  int n = lane & 15, grp = lane >> 4, base = grp * 8;
  long long node = nb + n;

  float f0[16], f1[16];
  const float* sp = s + node * 32 + grp * 16;
  const float* ap = s_agg + node * 32 + grp * 16;
#pragma unroll
  for (int k = 0; k < 16; ++k) { f0[k] = sp[k]; f1[k] = ap[k]; }
  v16h B0 = pack_b16(f0), B1 = pack_b16(f1);

  // phi_s layer0: 64->32 (LN+relu)
  v8f t0 = vzero(), t1 = vzero();
  t0 = wmma_f16(load_tileA(wpk, 14, lane), B0, t0);
  t1 = wmma_f16(load_tileA(wpk, 15, lane), B0, t1);
  t0 = wmma_f16(load_tileA(wpk, 16, lane), B1, t0);
  t1 = wmma_f16(load_tileA(wpk, 17, lane), B1, t1);
  add_bias(t0, t1, ps + PS_B0, base);
  ln_act(t0, t1, ps + PS_G0, ps + PS_BT0, base, true);

  // phi_s layer1: 32->32 (linear); residual
  v16h Bx = d_to_b(t0, t1, lane);
  v8f u0 = vzero(), u1 = vzero();
  u0 = wmma_f16(load_tileA(wpk, 18, lane), Bx, u0);
  u1 = wmma_f16(load_tileA(wpk, 19, lane), Bx, u1);
  add_bias(u0, u1, ps + PS_B1, base);
  float* os = out_s + node * 32;
  const float* srow = s + node * 32;
#pragma unroll
  for (int r = 0; r < 8; ++r) {
    os[base + r]      = srow[base + r]      + u0[r];
    os[16 + base + r] = srow[16 + base + r] + u1[r];
  }
  if (lane < 16) {
#pragma unroll
    for (int c = 0; c < 3; ++c)
      out_v[node * 3 + c] = v[node * 3 + c] + x_agg[node * 3 + c];
  }
}

extern "C" void kernel_launch(void* const* d_in, const int* in_sizes, int n_in,
                              void* d_out, int out_size, void* d_ws, size_t ws_size,
                              hipStream_t stream) {
  const float* v  = (const float*)d_in[0];
  const float* s  = (const float*)d_in[1];
  const float* pe = (const float*)d_in[2];
  const float* px = (const float*)d_in[3];
  const float* pm = (const float*)d_in[4];
  const float* ps = (const float*)d_in[5];
  const long long* ei = (const long long*)d_in[6];
  (void)in_sizes; (void)n_in; (void)out_size; (void)ws_size;

  char* ws = (char*)d_ws;
  unsigned int* wpk = (unsigned int*)(ws);
  float* x_agg = (float*)(ws + XAGG_OFF);
  float* s_agg = (float*)(ws + SAGG_OFF);

  float* out_v = (float*)d_out;                  // [N,3]
  float* out_s = out_v + (size_t)NN * 3;         // [N,32]
  float* out_e = out_s + (size_t)NN * 32;        // [E,32]

  // x_agg (3*N) and s_agg (32*N) are contiguous: zero both in one pass.
  EB_zero_kernel<<<512, 256, 0, stream>>>(x_agg, (long long)NN * 35);
  EB_pack_kernel<<<1, 256, 0, stream>>>(pe, px, pm, ps, wpk);
  EB_edge_kernel<<<NE / 128, 256, 0, stream>>>(v, s, pe, px, pm, ei, wpk,
                                               out_e, x_agg, s_agg);
  EB_node_kernel<<<(NN / 16 + 7) / 8, 256, 0, stream>>>(v, s, ps, wpk, x_agg,
                                                        s_agg, out_v, out_s);
}
